// HungarianMatcher_38448547233826
// MI455X (gfx1250) — matrix operand order repaired and checked by reading, not compiled, not run on hardware
//
#include <hip/hip_runtime.h>
#include <hip/hip_bf16.h>
#include <math.h>

#define B_  32
#define N_  128
#define D_  2048
#define H_  1024
#define M_  (B_ * N_)   // 4096 total rows

typedef _Float16 v16h __attribute__((ext_vector_type(16)));
typedef _Float16 v8h  __attribute__((ext_vector_type(8)));
typedef _Float16 v4h  __attribute__((ext_vector_type(4)));
typedef float    v8f  __attribute__((ext_vector_type(8)));
typedef float    v4f  __attribute__((ext_vector_type(4)));
typedef float    v2f  __attribute__((ext_vector_type(2)));
typedef unsigned int u32x4 __attribute__((ext_vector_type(4)));
typedef int          i32x4 __attribute__((ext_vector_type(4)));
typedef int          i32x8 __attribute__((ext_vector_type(8)));

// ---------------------------------------------------------------------------
// TDM: issue a 2-D tensor_load_to_lds. Descriptor per CDNA5 ISA §8.3/8.4.
// data_size = 2 bytes (f16). lds_addr = low 32 bits of generic shared ptr.
// ---------------------------------------------------------------------------
__device__ __forceinline__ void tdm_load_2d(unsigned lds_addr, const void* gaddr,
                                            unsigned tensor_d0, unsigned tensor_d1,
                                            unsigned stride0,
                                            unsigned tile0, unsigned tile1) {
  unsigned long long ga = (unsigned long long)gaddr;
  u32x4 g0;
  g0[0] = 1u;                                    // count=1 (valid), user mode
  g0[1] = lds_addr;                              // LDS byte address
  g0[2] = (unsigned)(ga & 0xFFFFFFFFu);          // global_addr[31:0]
  g0[3] = (unsigned)((ga >> 32) & 0x01FFFFFFu)   // global_addr[56:32]
          | (2u << 30);                          // type = 2 ("image")
  i32x8 g1;
  g1[0] = (int)(1u << 16);                       // data_size=1 (2B), wg_mask=0
  g1[1] = (int)((tensor_d0 & 0xFFFFu) << 16);    // tensor_dim0[15:0] @bits63:48
  g1[2] = (int)((tensor_d0 >> 16) | ((tensor_d1 & 0xFFFFu) << 16));
  g1[3] = (int)((tensor_d1 >> 16) | (tile0 << 16));        // tile_dim0 @127:112
  g1[4] = (int)(tile1 & 0xFFFFu);                // tile_dim1; tile_dim2=0
  g1[5] = (int)stride0;                          // tensor_dim0_stride[31:0]
  g1[6] = 0;                                     // stride hi + dim1_stride lo
  g1[7] = 0;
  i32x4 z4 = {0, 0, 0, 0};
#if defined(__clang_major__) && __clang_major__ >= 23
  i32x8 z8 = {0, 0, 0, 0, 0, 0, 0, 0};
  __builtin_amdgcn_tensor_load_to_lds(g0, g1, z4, z4, z8, 0);
#else
  __builtin_amdgcn_tensor_load_to_lds(g0, g1, z4, z4, 0);
#endif
}

// ---------------------------------------------------------------------------
// Row norms: one wave per row (support rows then query rows).
// ---------------------------------------------------------------------------
__global__ __launch_bounds__(256) void norms_kernel(
    const float* __restrict__ S, const float* __restrict__ Q,
    float* __restrict__ ns, float* __restrict__ nq) {
  int wave = (blockIdx.x * blockDim.x + threadIdx.x) >> 5;
  int lane = threadIdx.x & 31;
  const float* src;
  float* dst;
  int row;
  if (wave < M_) { src = S; dst = ns; row = wave; }
  else           { src = Q; dst = nq; row = wave - M_; }
  const float* p = src + (size_t)row * D_;
  float acc = 0.f;
  for (int d = lane; d < D_; d += 32) { float x = p[d]; acc += x * x; }
  for (int off = 16; off > 0; off >>= 1) acc += __shfl_xor(acc, off, 32);
  if (lane == 0) dst[row] = sqrtf(acc);
}

// ---------------------------------------------------------------------------
// Cost matrix via fp32 WMMA (V_WMMA_F32_16X16X4_F32) -- full fp32 since the
// Hungarian assignment is tie-sensitive. One wave per 16x16 tile.
// ---------------------------------------------------------------------------
__global__ __launch_bounds__(32) void cost_kernel(
    const float* __restrict__ S, const float* __restrict__ Q,
    const float* __restrict__ ns, const float* __restrict__ nq,
    float* __restrict__ cost) {
  int bid = blockIdx.x;               // b*64 + tile
  int b = bid >> 6, tile = bid & 63;
  int tm = (tile >> 3) << 4, tn = (tile & 7) << 4;
  int lane = threadIdx.x & 31;
  int hv = lane >> 4, l = lane & 15;
  const float* arow = S + ((size_t)b * N_ + tm + l) * D_ + 2 * hv;
  const float* brow = Q + ((size_t)b * N_ + tn + l) * D_ + 2 * hv;
  v8f acc = {0.f, 0.f, 0.f, 0.f, 0.f, 0.f, 0.f, 0.f};
  for (int k0 = 0; k0 < D_; k0 += 4) {
    v2f a  = *(const v2f*)(arow + k0);
    v2f bb = *(const v2f*)(brow + k0);
    acc = __builtin_amdgcn_wmma_f32_16x16x4_f32(
        false, a, false, bb, (short)0, acc, false, false);
  }
  float qn = nq[b * N_ + tn + l];
#pragma unroll
  for (int v = 0; v < 8; ++v) {
    int m = tm + v + 8 * hv;
    cost[((size_t)b * N_ + m) * N_ + tn + l] =
        1.0f - acc[v] / (ns[b * N_ + m] * qn);
  }
}

// ---------------------------------------------------------------------------
// Jonker-Volgenant LAP, one 128-thread block per batch, float64 duals to
// match the reference solver (first-index tie-break like np.argmin).
// ---------------------------------------------------------------------------
__global__ __launch_bounds__(128) void lap_kernel(
    const float* __restrict__ cost, int* __restrict__ qi) {
  int b = blockIdx.x, tid = threadIdx.x;
  __shared__ double u[N_ + 1], vv[N_ + 1], minv[N_ + 1];
  __shared__ int p[N_ + 1], way[N_ + 1], used[N_ + 1];
  __shared__ double redv[N_];
  __shared__ int redj[N_];
  const float* C = cost + (size_t)b * N_ * N_;
  const double INF = __builtin_huge_val();

  for (int j = tid; j <= N_; j += N_) { u[j] = 0.0; vv[j] = 0.0; p[j] = 0; way[j] = 0; }
  __syncthreads();

  for (int i = 1; i <= N_; ++i) {
    minv[tid] = INF; used[tid] = 0;
    if (tid == 0) { minv[N_] = INF; used[N_] = 0; p[0] = i; }
    __syncthreads();
    int j0 = 0;
    for (;;) {
      if (tid == 0) used[j0] = 1;
      __syncthreads();
      int i0 = p[j0];
      int j = tid + 1;                 // columns 1..128
      double mv;
      if (!used[j]) {
        double c = (double)C[(i0 - 1) * N_ + (j - 1)] - u[i0] - vv[j];
        if (c < minv[j]) { minv[j] = c; way[j] = j0; }
        mv = minv[j];
      } else {
        mv = INF;                      // unused cols always have finite minv
      }
      redv[tid] = mv; redj[tid] = j;
      __syncthreads();
      for (int s = 64; s > 0; s >>= 1) {
        if (tid < s) {
          double v2 = redv[tid + s]; int j2 = redj[tid + s];
          if (v2 < redv[tid] || (v2 == redv[tid] && j2 < redj[tid])) {
            redv[tid] = v2; redj[tid] = j2;
          }
        }
        __syncthreads();
      }
      int j1 = redj[0];
      double delta = redv[0];
      __syncthreads();
      if (used[j]) { u[p[j]] += delta; vv[j] -= delta; }
      else         { minv[j] -= delta; }
      if (tid == 0 && used[0]) { u[p[0]] += delta; vv[0] -= delta; }
      __syncthreads();
      j0 = j1;
      if (p[j0] == 0) break;           // p unchanged inside loop: uniform
    }
    if (tid == 0) {                    // augment along "way"
      int ja = j0;
      while (ja) { int jb = way[ja]; p[ja] = p[jb]; ja = jb; }
    }
    __syncthreads();
  }
  int j = tid + 1;
  qi[b * N_ + (p[j] - 1)] = j - 1;     // col_of_row
}

// ---------------------------------------------------------------------------
// fp32 -> f16 elementwise conversion (n multiple of 4)
// ---------------------------------------------------------------------------
__global__ __launch_bounds__(256) void convert_f16_kernel(
    const float* __restrict__ src, _Float16* __restrict__ dst, int n) {
  int idx = (blockIdx.x * blockDim.x + threadIdx.x) * 4;
  if (idx < n) {
    v4f x = *(const v4f*)(src + idx);
    v4h y = {(_Float16)x.x, (_Float16)x.y, (_Float16)x.z, (_Float16)x.w};
    *(v4h*)(dst + idx) = y;
  }
}

__global__ __launch_bounds__(256) void rowmap_s_kernel(int* __restrict__ rowmap) {
  int m = blockIdx.x * blockDim.x + threadIdx.x;
  if (m < M_) rowmap[m] = m;
}

// Gather matched query rows -> f16 X buffer, and record source rows.
__global__ __launch_bounds__(256) void gather_q_kernel(
    const float* __restrict__ Q, const int* __restrict__ qi,
    _Float16* __restrict__ Xq, int* __restrict__ rowmap) {
  int m = blockIdx.x;                  // 0..4095
  int b = m >> 7;
  int src = b * N_ + qi[m];
  if (threadIdx.x == 0) rowmap[m] = src;
  const float* s = Q + (size_t)src * D_;
  _Float16* d = Xq + (size_t)m * D_;
  for (int idx = threadIdx.x * 4; idx < D_; idx += blockDim.x * 4) {
    v4f x = *(const v4f*)(s + idx);
    v4h y = {(_Float16)x.x, (_Float16)x.y, (_Float16)x.z, (_Float16)x.w};
    *(v4h*)(d + idx) = y;
  }
}

// ---------------------------------------------------------------------------
// TDM-staged, double-buffered block GEMM. 256 threads (8 waves) per block,
// 64x64 macro-tile, KC=64 K-chunks. Wave 0 drives the Tensor Data Mover:
// issues tensor_load_to_lds for the next chunk's A/B tiles, all waves compute
// the current chunk from LDS (ds_load_b128 fragments -> v_wmma_f32_16x16x32_f16),
// wave 0 waits TENSORcnt, block barriers, buffers swap.
// Wave w computes sub-tiles (msub = w&3, nsub = (w>>2)*2 + {0,1}).
// ---------------------------------------------------------------------------
#define KC_ 64

__device__ __forceinline__ v16h frag_a(const _Float16* Ac, int row, int k, int hv) {
  const _Float16* ap = Ac + row * KC_ + k + 8 * hv;   // K blocks {b..b+7, b+16..b+23}
  v8h lo = *(const v8h*)ap;
  v8h hi = *(const v8h*)(ap + 16);
  return __builtin_shufflevector(lo, hi, 0,1,2,3,4,5,6,7,8,9,10,11,12,13,14,15);
}
__device__ __forceinline__ v16h frag_b(const _Float16* Bc, int row, int k, int hv) {
  const _Float16* bp = Bc + row * KC_ + k + 8 * hv;   // K blocks {b..b+7, b+16..b+23}
  v8h lo = *(const v8h*)bp;
  v8h hi = *(const v8h*)(bp + 16);
  return __builtin_shufflevector(lo, hi, 0,1,2,3,4,5,6,7,8,9,10,11,12,13,14,15);
}

// GEMM1: Hout = relu(X @ W^T), f16 in/out, f32 accumulate.
__global__ __launch_bounds__(256) void gemm1_kernel(
    const _Float16* __restrict__ X, const _Float16* __restrict__ W,
    _Float16* __restrict__ Hout, int K, int Nout) {
  __shared__ __align__(16) _Float16 As[2][64 * KC_];
  __shared__ __align__(16) _Float16 Bs[2][64 * KC_];
  int tilesN = Nout >> 6;
  int tm = (blockIdx.x / tilesN) << 6;
  int tn = (blockIdx.x % tilesN) << 6;
  int tid = threadIdx.x;
  int wave = tid >> 5, lane = tid & 31;
  int hv = lane >> 4, l = lane & 15;
  int msub = wave & 3;
  int nsub = (wave >> 2) << 1;
  unsigned aLds = (unsigned)(size_t)&As[0][0];
  unsigned bLds = (unsigned)(size_t)&Bs[0][0];
  const unsigned bufBytes = 64 * KC_ * 2;
  v8f acc0 = {0.f,0.f,0.f,0.f,0.f,0.f,0.f,0.f};
  v8f acc1 = {0.f,0.f,0.f,0.f,0.f,0.f,0.f,0.f};
  int nChunks = K / KC_;
  if (wave == 0) {
    tdm_load_2d(aLds, X + (size_t)tm * K, (unsigned)K, 64u, (unsigned)K, KC_, 64u);
    tdm_load_2d(bLds, W + (size_t)tn * K, (unsigned)K, 64u, (unsigned)K, KC_, 64u);
    __builtin_amdgcn_s_wait_tensorcnt(0);
  }
  __syncthreads();
  for (int c = 0; c < nChunks; ++c) {
    int cur = c & 1, nxt = cur ^ 1;
    if (c + 1 < nChunks && wave == 0) {
      size_t k0 = (size_t)(c + 1) * KC_;
      tdm_load_2d(aLds + nxt * bufBytes, X + (size_t)tm * K + k0,
                  (unsigned)K, 64u, (unsigned)K, KC_, 64u);
      tdm_load_2d(bLds + nxt * bufBytes, W + (size_t)tn * K + k0,
                  (unsigned)K, 64u, (unsigned)K, KC_, 64u);
    }
    const _Float16* Ac = &As[cur][0];
    const _Float16* Bc = &Bs[cur][0];
#pragma unroll
    for (int k = 0; k < KC_; k += 32) {
      v16h a  = frag_a(Ac, msub * 16 + l, k, hv);
      v16h b0 = frag_b(Bc, nsub * 16 + l, k, hv);
      acc0 = __builtin_amdgcn_wmma_f32_16x16x32_f16(
          false, a, false, b0, (short)0, acc0, false, false);
      v16h b1 = frag_b(Bc, (nsub + 1) * 16 + l, k, hv);
      acc1 = __builtin_amdgcn_wmma_f32_16x16x32_f16(
          false, a, false, b1, (short)0, acc1, false, false);
    }
    if (wave == 0) __builtin_amdgcn_s_wait_tensorcnt(0);
    __syncthreads();
  }
#pragma unroll
  for (int v = 0; v < 8; ++v) {
    int m = tm + msub * 16 + v + 8 * hv;
    float r0 = acc0[v] > 0.f ? acc0[v] : 0.f;
    float r1 = acc1[v] > 0.f ? acc1[v] : 0.f;
    Hout[(size_t)m * Nout + tn + nsub * 16 + l]       = (_Float16)r0;
    Hout[(size_t)m * Nout + tn + (nsub + 1) * 16 + l] = (_Float16)r1;
  }
}

// GEMM2: out = resid[rowmap[m]] + relu(H @ W2^T), f32 out. K=1024, Nout=2048.
__global__ __launch_bounds__(256) void gemm2_kernel(
    const _Float16* __restrict__ Hin, const _Float16* __restrict__ W,
    const float* __restrict__ resid, const int* __restrict__ rowmap,
    float* __restrict__ out) {
  const int K = H_, Nout = D_;
  __shared__ __align__(16) _Float16 As[2][64 * KC_];
  __shared__ __align__(16) _Float16 Bs[2][64 * KC_];
  int tilesN = Nout >> 6;              // 32
  int tm = (blockIdx.x / tilesN) << 6;
  int tn = (blockIdx.x % tilesN) << 6;
  int tid = threadIdx.x;
  int wave = tid >> 5, lane = tid & 31;
  int hv = lane >> 4, l = lane & 15;
  int msub = wave & 3;
  int nsub = (wave >> 2) << 1;
  unsigned aLds = (unsigned)(size_t)&As[0][0];
  unsigned bLds = (unsigned)(size_t)&Bs[0][0];
  const unsigned bufBytes = 64 * KC_ * 2;
  v8f acc0 = {0.f,0.f,0.f,0.f,0.f,0.f,0.f,0.f};
  v8f acc1 = {0.f,0.f,0.f,0.f,0.f,0.f,0.f,0.f};
  int nChunks = K / KC_;               // 16
  if (wave == 0) {
    tdm_load_2d(aLds, Hin + (size_t)tm * K, (unsigned)K, 64u, (unsigned)K, KC_, 64u);
    tdm_load_2d(bLds, W + (size_t)tn * K, (unsigned)K, 64u, (unsigned)K, KC_, 64u);
    __builtin_amdgcn_s_wait_tensorcnt(0);
  }
  __syncthreads();
  for (int c = 0; c < nChunks; ++c) {
    int cur = c & 1, nxt = cur ^ 1;
    if (c + 1 < nChunks && wave == 0) {
      size_t k0 = (size_t)(c + 1) * KC_;
      tdm_load_2d(aLds + nxt * bufBytes, Hin + (size_t)tm * K + k0,
                  (unsigned)K, 64u, (unsigned)K, KC_, 64u);
      tdm_load_2d(bLds + nxt * bufBytes, W + (size_t)tn * K + k0,
                  (unsigned)K, 64u, (unsigned)K, KC_, 64u);
    }
    const _Float16* Ac = &As[cur][0];
    const _Float16* Bc = &Bs[cur][0];
#pragma unroll
    for (int k = 0; k < KC_; k += 32) {
      v16h a  = frag_a(Ac, msub * 16 + l, k, hv);
      v16h b0 = frag_b(Bc, nsub * 16 + l, k, hv);
      acc0 = __builtin_amdgcn_wmma_f32_16x16x32_f16(
          false, a, false, b0, (short)0, acc0, false, false);
      v16h b1 = frag_b(Bc, (nsub + 1) * 16 + l, k, hv);
      acc1 = __builtin_amdgcn_wmma_f32_16x16x32_f16(
          false, a, false, b1, (short)0, acc1, false, false);
    }
    if (wave == 0) __builtin_amdgcn_s_wait_tensorcnt(0);
    __syncthreads();
  }
#pragma unroll
  for (int v = 0; v < 8; ++v) {
    int m = tm + msub * 16 + v + 8 * hv;
    int src = rowmap[m];
    float r0 = acc0[v] > 0.f ? acc0[v] : 0.f;
    float r1 = acc1[v] > 0.f ? acc1[v] : 0.f;
    int n0 = tn + nsub * 16 + l;
    int n1 = tn + (nsub + 1) * 16 + l;
    out[(size_t)m * Nout + n0] = resid[(size_t)src * Nout + n0] + r0;
    out[(size_t)m * Nout + n1] = resid[(size_t)src * Nout + n1] + r1;
  }
}

// ---------------------------------------------------------------------------
extern "C" void kernel_launch(void* const* d_in, const int* in_sizes, int n_in,
                              void* d_out, int out_size, void* d_ws, size_t ws_size,
                              hipStream_t stream) {
  const float* support = (const float*)d_in[0];
  const float* query   = (const float*)d_in[1];
  const float* W1      = (const float*)d_in[2];   // [1024, 2048]
  const float* W2      = (const float*)d_in[3];   // [2048, 1024]
  float* out0 = (float*)d_out;
  float* out1 = out0 + (size_t)M_ * D_;

  char* ws = (char*)d_ws;
  size_t off = 0;
  auto alloc = [&](size_t bytes) -> char* {
    char* p = ws + off;
    off = (off + bytes + 255) & ~(size_t)255;
    return p;
  };
  float*    cost    = (float*)alloc((size_t)B_ * N_ * N_ * 4);   // 2 MB
  int*      qi      = (int*)alloc((size_t)M_ * 4);
  float*    ns      = (float*)alloc((size_t)M_ * 4);
  float*    nq      = (float*)alloc((size_t)M_ * 4);
  int*      rowmapS = (int*)alloc((size_t)M_ * 4);
  int*      rowmapQ = (int*)alloc((size_t)M_ * 4);
  _Float16* W1h     = (_Float16*)alloc((size_t)D_ * H_ * 2);     // 4 MB
  _Float16* W2h     = (_Float16*)alloc((size_t)D_ * H_ * 2);     // 4 MB
  _Float16* Xbuf    = (_Float16*)alloc((size_t)M_ * D_ * 2);     // 16 MB (reused s/q)
  _Float16* Hbuf    = (_Float16*)alloc((size_t)M_ * H_ * 2);     // 8 MB (reused s/q)

  // 1) norms + weight conversion
  norms_kernel<<<(2 * M_ * 32) / 256, 256, 0, stream>>>(support, query, ns, nq);
  convert_f16_kernel<<<(D_ * H_ / 4 + 255) / 256, 256, 0, stream>>>(W1, W1h, D_ * H_);
  convert_f16_kernel<<<(D_ * H_ / 4 + 255) / 256, 256, 0, stream>>>(W2, W2h, D_ * H_);

  // 2) fp32 cosine-cost via f32 WMMA, then Hungarian assignment per batch
  cost_kernel<<<B_ * 64, 32, 0, stream>>>(support, query, ns, nq, cost);
  lap_kernel<<<B_, 128, 0, stream>>>(cost, qi);

  // 3) support path: out0 = support + relu(relu(support@W1^T)@W2^T)
  convert_f16_kernel<<<(M_ * D_ / 4 + 255) / 256, 256, 0, stream>>>(support, Xbuf, M_ * D_);
  rowmap_s_kernel<<<M_ / 256, 256, 0, stream>>>(rowmapS);
  gemm1_kernel<<<(M_ / 64) * (H_ / 64), 256, 0, stream>>>(Xbuf, W1h, Hbuf, D_, H_);
  gemm2_kernel<<<(M_ / 64) * (D_ / 64), 256, 0, stream>>>(Hbuf, W2h, support, rowmapS, out0);

  // 4) query path: gather matched rows, then same MLP + residual
  gather_q_kernel<<<M_, 256, 0, stream>>>(query, qi, Xbuf, rowmapQ);
  gemm1_kernel<<<(M_ / 64) * (H_ / 64), 256, 0, stream>>>(Xbuf, W1h, Hbuf, D_, H_);
  gemm2_kernel<<<(M_ / 64) * (D_ / 64), 256, 0, stream>>>(Hbuf, W2h, query, rowmapQ, out1);
}
